// TMNRR_12987981103182
// MI455X (gfx1250) — matrix operand order repaired and checked by reading, not compiled, not run on hardware
//
#include <hip/hip_runtime.h>
#include <hip/hip_bf16.h>

// ---------------------------------------------------------------- constants
static constexpr int VV  = 4;
static constexpr int Bsz = 4096;
static constexpr int Dd  = 1024;
static constexpr int Hh  = 512;
static constexpr int Cc  = 50;
static constexpr int SAMPLE_NUM = 10000;

// ---------------------------------------------------------------- types
typedef __attribute__((ext_vector_type(16))) __bf16 bf16x16;
typedef __attribute__((ext_vector_type(8)))  __bf16 bf16x8;
typedef __attribute__((ext_vector_type(8)))  float  v8f;

// ---------------------------------------------------------------- helpers
__device__ __forceinline__ unsigned short bfbits(float f) {
    unsigned u = __builtin_bit_cast(unsigned, f);
    u += 0x7FFFu + ((u >> 16) & 1u);           // round-to-nearest-even
    return (unsigned short)(u >> 16);
}
__device__ __forceinline__ unsigned pack2bf(float a, float b) {
    return (unsigned)bfbits(a) | ((unsigned)bfbits(b) << 16);
}
__device__ __forceinline__ __bf16 f2bf(float f) {
    unsigned short h = bfbits(f);
    return __builtin_bit_cast(__bf16, h);
}
__device__ __forceinline__ float softplusf(float x) {
    float ax = fabsf(x);
    return __logf(1.f + __expf(-ax)) + fmaxf(x, 0.f);
}
__device__ __forceinline__ float wave_sum(float x) {
    #pragma unroll
    for (int o = 16; o > 0; o >>= 1) x += __shfl_xor(x, o, 32);
    return x;
}
__device__ __forceinline__ v8f v8f_zero() {
    v8f z = {0.f, 0.f, 0.f, 0.f, 0.f, 0.f, 0.f, 0.f};
    return z;
}
// Build a 16x32 bf16 A-fragment: lane L holds row (L&15); low 8 bf16 at
// k = (L<16?0:16), high 8 bf16 at k+8 (16 consecutive k, matching B).
__device__ __forceinline__ bf16x16 load_a_frag(const __bf16* lds, int rowBase,
                                               int lane, int ldk) {
    int row = rowBase + (lane & 15);
    int kb  = (lane < 16) ? 0 : 16;
    bf16x8 lo = *(const bf16x8*)(lds + row * ldk + kb);
    bf16x8 hi = *(const bf16x8*)(lds + row * ldk + kb + 8);
    bf16x16 a;
    #pragma unroll
    for (int i = 0; i < 8; ++i) { a[i] = lo[i]; a[8 + i] = hi[i]; }
    return a;
}
// Build a 32x16 bf16 B-fragment from N-major LDS [n][k]: lane L holds col
// (L&15), 16 consecutive k starting at (L<16?0:16).
__device__ __forceinline__ bf16x16 load_b_frag(const __bf16* lds, int colBase,
                                               int lane, int ldk) {
    int col = colBase + (lane & 15);
    int kb  = (lane < 16) ? 0 : 16;
    return *(const bf16x16*)(lds + col * ldk + kb);
}

// ================================================================ kernel 1
// h[v,m,n] = relu( sum_k x[v,m,k]*W1[v,k,n] + b1[v,n] )   (bf16 out)
// Tile: 128(M) x 64(N) x 32(K). 8 waves (4x2), each wave 2x2 WMMA tiles.
__global__ __launch_bounds__(256)
void gemm1_relu(const float* __restrict__ x, const float* __restrict__ W1,
                const float* __restrict__ b1, __bf16* __restrict__ hout) {
    __shared__ __bf16 lA[128 * 32];
    __shared__ __bf16 lB[64 * 32];
    const int v   = blockIdx.z;
    const int m0  = blockIdx.x * 128;
    const int n0  = blockIdx.y * 64;
    const int tid = threadIdx.x;
    const int lane = tid & 31, wave = tid >> 5;
    const int wm = wave & 3, wn = wave >> 2;          // 4 x 2 wave grid

    const float* xA = x  + (size_t)v * Bsz * Dd + (size_t)m0 * Dd;
    const float* wB = W1 + (size_t)v * Dd * Hh + n0;

    v8f acc[2][2];
    #pragma unroll
    for (int i = 0; i < 2; ++i)
        #pragma unroll
        for (int j = 0; j < 2; ++j) acc[i][j] = v8f_zero();

    for (int k0 = 0; k0 < Dd; k0 += 32) {
        __syncthreads();
        // stage A: 128x32 f32 -> bf16, 4 x float4 per thread
        #pragma unroll
        for (int j = 0; j < 4; ++j) {
            int e = (tid + 256 * j) * 4;
            int row = e >> 5, col = e & 31;
            float4 f = *(const float4*)(xA + (size_t)row * Dd + k0 + col);
            uint2 p;
            p.x = pack2bf(f.x, f.y);
            p.y = pack2bf(f.z, f.w);
            *(uint2*)(lA + row * 32 + col) = p;
        }
        // stage B: 32x64 f32 -> bf16, stored N-major [n][k]
        #pragma unroll
        for (int j = 0; j < 2; ++j) {
            int e = (tid + 256 * j) * 4;
            int k = e >> 6, n = e & 63;
            float4 f = *(const float4*)(wB + (size_t)(k0 + k) * Hh + n);
            lB[(n + 0) * 32 + k] = f2bf(f.x);
            lB[(n + 1) * 32 + k] = f2bf(f.y);
            lB[(n + 2) * 32 + k] = f2bf(f.z);
            lB[(n + 3) * 32 + k] = f2bf(f.w);
        }
        __syncthreads();

        bf16x16 af[2], bf[2];
        #pragma unroll
        for (int i = 0; i < 2; ++i) af[i] = load_a_frag(lA, wm * 32 + i * 16, lane, 32);
        #pragma unroll
        for (int i = 0; i < 2; ++i) bf[i] = load_b_frag(lB, wn * 32 + i * 16, lane, 32);
        #pragma unroll
        for (int mi = 0; mi < 2; ++mi)
            #pragma unroll
            for (int ni = 0; ni < 2; ++ni)
                acc[mi][ni] = __builtin_amdgcn_wmma_f32_16x16x32_bf16(
                    false, af[mi], false, bf[ni], (short)0, acc[mi][ni], false, false);
    }

    // epilogue: C/D layout -> VGPR j holds M=j (lanes 0-15) / M=j+8 (16-31)
    const int rext = (lane >> 4) * 8;
    const int ncol = lane & 15;
    #pragma unroll
    for (int mi = 0; mi < 2; ++mi)
        #pragma unroll
        for (int ni = 0; ni < 2; ++ni) {
            int gn = n0 + wn * 32 + ni * 16 + ncol;
            float bias = b1[v * Hh + gn];
            #pragma unroll
            for (int j = 0; j < 8; ++j) {
                int gm = m0 + wm * 32 + mi * 16 + j + rext;
                float val = fmaxf(acc[mi][ni][j] + bias, 0.f);
                hout[(size_t)v * Bsz * Hh + (size_t)gm * Hh + gn] = f2bf(val);
            }
        }
}

// ================================================================ kernel 2
// evidence[v,m,c] = softplus( sum_k h[v,m,k]*W2[v,k,c] + b2[v,c] )
// Tile: 128(M) x 64(N, C=50 padded) x 32(K).
__global__ __launch_bounds__(256)
void gemm2_softplus(const __bf16* __restrict__ hin, const float* __restrict__ W2,
                    const float* __restrict__ b2, float* __restrict__ evidence) {
    __shared__ __bf16 lA[128 * 32];
    __shared__ __bf16 lB[64 * 32];
    const int v   = blockIdx.z;
    const int m0  = blockIdx.x * 128;
    const int tid = threadIdx.x;
    const int lane = tid & 31, wave = tid >> 5;
    const int wm = wave & 3, wn = wave >> 2;

    const __bf16* hA = hin + (size_t)v * Bsz * Hh + (size_t)m0 * Hh;
    const float*  wB = W2  + (size_t)v * Hh * Cc;

    v8f acc[2][2];
    #pragma unroll
    for (int i = 0; i < 2; ++i)
        #pragma unroll
        for (int j = 0; j < 2; ++j) acc[i][j] = v8f_zero();

    for (int k0 = 0; k0 < Hh; k0 += 32) {
        __syncthreads();
        // stage A: already bf16, 2 x 16-byte copies per thread
        #pragma unroll
        for (int j = 0; j < 2; ++j) {
            int e = (tid + 256 * j) * 8;
            int row = e >> 5, col = e & 31;
            *(uint4*)(lA + row * 32 + col) =
                *(const uint4*)(hA + (size_t)row * Hh + k0 + col);
        }
        // stage B: 32 x 64 (cols >= 50 zero-filled), N-major
        #pragma unroll
        for (int j = 0; j < 8; ++j) {
            int e = tid + 256 * j;
            int k = e >> 6, n = e & 63;
            float val = (n < Cc) ? wB[(size_t)(k0 + k) * Cc + n] : 0.f;
            lB[n * 32 + k] = f2bf(val);
        }
        __syncthreads();

        bf16x16 af[2], bf[2];
        #pragma unroll
        for (int i = 0; i < 2; ++i) af[i] = load_a_frag(lA, wm * 32 + i * 16, lane, 32);
        #pragma unroll
        for (int i = 0; i < 2; ++i) bf[i] = load_b_frag(lB, wn * 32 + i * 16, lane, 32);
        #pragma unroll
        for (int mi = 0; mi < 2; ++mi)
            #pragma unroll
            for (int ni = 0; ni < 2; ++ni)
                acc[mi][ni] = __builtin_amdgcn_wmma_f32_16x16x32_bf16(
                    false, af[mi], false, bf[ni], (short)0, acc[mi][ni], false, false);
    }

    const int rext = (lane >> 4) * 8;
    const int ncol = lane & 15;
    #pragma unroll
    for (int mi = 0; mi < 2; ++mi)
        #pragma unroll
        for (int ni = 0; ni < 2; ++ni) {
            int gn = wn * 32 + ni * 16 + ncol;
            if (gn >= Cc) continue;
            float bias = b2[v * Cc + gn];
            #pragma unroll
            for (int j = 0; j < 8; ++j) {
                int gm = m0 + wm * 32 + mi * 16 + j + rext;
                float val = softplusf(acc[mi][ni][j] + bias);
                evidence[((size_t)v * Bsz + gm) * Cc + gn] = val;
            }
        }
}

// ================================================================ kernel 3
// transferred[v,b,d] = sum_c evidence[v,b,c] * matrixes[v, indexes[b], c, d]
// one wave per (v,b); lanes own columns d and d+32.
__global__ __launch_bounds__(256)
void transfer_kernel(const float* __restrict__ evidence,
                     const int* __restrict__ indexes,
                     const float* __restrict__ matrixes,
                     float* __restrict__ transferred) {
    int wid  = (int)((blockIdx.x * blockDim.x + threadIdx.x) >> 5);
    int lane = threadIdx.x & 31;
    if (wid >= VV * Bsz) return;
    int v = wid >> 12;                  // Bsz == 4096
    int b = wid & (Bsz - 1);
    int idx = indexes[b];
    const float* M = matrixes + ((size_t)v * SAMPLE_NUM + idx) * Cc * Cc;
    const float* e = evidence + ((size_t)v * Bsz + b) * Cc;
    const int d0 = lane, d1 = lane + 32;
    float acc0 = 0.f, acc1 = 0.f;
    for (int c = 0; c < Cc; ++c) {
        float ec = e[c];
        const float* row = M + c * Cc;
        __builtin_prefetch(row + Cc, 0, 0);
        acc0 += ec * row[d0];
        if (d1 < Cc) acc1 += ec * row[d1];
    }
    float* t = transferred + ((size_t)v * Bsz + b) * Cc;
    t[d0] = acc0;
    if (d1 < Cc) t[d1] = acc1;
}

// ================================================================ kernel 4
// Sequential Dempster-Shafer combination of the 4 views; one wave per b.
__global__ __launch_bounds__(256)
void ds_combine_kernel(const float* __restrict__ transferred,
                       float* __restrict__ evidence_a) {
    int wid  = (int)((blockIdx.x * blockDim.x + threadIdx.x) >> 5);
    int lane = threadIdx.x & 31;
    if (wid >= Bsz) return;
    const int b = wid;
    const int c1 = lane + 32;
    const bool ok1 = (c1 < Cc);
    const float Kf = (float)Cc;

    const float* t0p = transferred + (size_t)b * Cc;
    float cur0 = t0p[lane] + 1.f;
    float cur1 = ok1 ? (t0p[c1] + 1.f) : 0.f;

    for (int v = 1; v < VV; ++v) {
        const float* tp = transferred + ((size_t)v * Bsz + b) * Cc;
        float t0 = tp[lane] + 1.f;
        float t1 = ok1 ? (tp[c1] + 1.f) : 0.f;

        float S1 = wave_sum(cur0 + cur1);
        float S2 = wave_sum(t0 + t1);
        float b10 = (cur0 - 1.f) / S1;
        float b11 = ok1 ? (cur1 - 1.f) / S1 : 0.f;
        float b20 = (t0 - 1.f) / S2;
        float b21 = ok1 ? (t1 - 1.f) / S2 : 0.f;
        float u1 = Kf / S1, u2 = Kf / S2;
        float sb1 = wave_sum(b10 + b11);
        float sb2 = wave_sum(b20 + b21);
        float sbb = wave_sum(b10 * b20 + b11 * b21);
        float inv = 1.f / (1.f - (sb1 * sb2 - sbb));
        float u_a = u1 * u2 * inv;
        float S_a = Kf / u_a;
        cur0 = (b10 * b20 + b10 * u2 + b20 * u1) * inv * S_a + 1.f;
        cur1 = ok1 ? (b11 * b21 + b11 * u2 + b21 * u1) * inv * S_a + 1.f : 0.f;
    }
    float* out = evidence_a + (size_t)b * Cc;
    out[lane] = cur0 - 1.f;
    if (ok1) out[c1] = cur1 - 1.f;
}

// ================================================================ launcher
extern "C" void kernel_launch(void* const* d_in, const int* in_sizes, int n_in,
                              void* d_out, int out_size, void* d_ws, size_t ws_size,
                              hipStream_t stream) {
    const float* x        = (const float*)d_in[0];
    const int*   indexes  = (const int*)  d_in[1];
    const float* W1       = (const float*)d_in[2];
    const float* b1       = (const float*)d_in[3];
    const float* W2       = (const float*)d_in[4];
    const float* b2       = (const float*)d_in[5];
    const float* matrixes = (const float*)d_in[6];

    float* out         = (float*)d_out;
    float* transferred = out;                                   // V*B*C
    float* evidence_a  = out + (size_t)VV * Bsz * Cc;           // B*C
    float* evidence    = evidence_a + (size_t)Bsz * Cc;         // V*B*C
    __bf16* hbuf       = (__bf16*)d_ws;                         // V*B*H bf16

    dim3 g1(Bsz / 128, Hh / 64, VV);   // 32 x 8 x 4 = 1024 blocks
    gemm1_relu<<<g1, 256, 0, stream>>>(x, W1, b1, hbuf);

    dim3 g2(Bsz / 128, 1, VV);         // 128 blocks
    gemm2_softplus<<<g2, 256, 0, stream>>>(hbuf, W2, b2, evidence);

    transfer_kernel<<<(VV * Bsz) / 8, 256, 0, stream>>>(evidence, indexes,
                                                        matrixes, transferred);

    ds_combine_kernel<<<Bsz / 8, 256, 0, stream>>>(transferred, evidence_a);
}
